// Model_40827959115911
// MI455X (gfx1250) — matrix operand order repaired and checked by reading, not compiled, not run on hardware
//
#include <hip/hip_runtime.h>
#include <hip/hip_bf16.h>
#include <math.h>

// ---------------------------------------------------------------------------
// CDNA5 / gfx1250 WMMA types & helpers (wave32)
// ---------------------------------------------------------------------------
typedef __attribute__((ext_vector_type(16))) _Float16 v16h;
typedef __attribute__((ext_vector_type(8)))  _Float16 v8h;
typedef __attribute__((ext_vector_type(8)))  float    v8f;

#define WMMA_F16(a, b, c) \
  __builtin_amdgcn_wmma_f32_16x16x32_f16(false, (a), false, (b), (short)0, (c), false, false)

// A fragment (16x32 f16 tile of row-major MxK source).
// contiguous k order used here (not the ISA table order): lane holds row m=lane&15; lanes 0-15 cover K {kh=0}, 16-31 {kh=1};
// elements 0..15 = K k0+kh*16.. contiguous -> two adjacent 16B runs.
static __device__ inline v16h load_a_frag(const _Float16* A, int lda, int M,
                                          int row0, int k0) {
  int lane = threadIdx.x & 31;
  int m = row0 + (lane & 15);
  if (m >= M) m = M - 1;                  // clamp; garbage rows never stored
  int kh = (lane >> 4) & 1;
  const _Float16* p = A + (size_t)m * lda + k0 + kh * 16;
  v8h lo = *(const v8h*)(p);
  v8h hi = *(const v8h*)(p + 8);
  return __builtin_shufflevector(lo, hi, 0, 1, 2, 3, 4, 5, 6, 7,
                                 8, 9, 10, 11, 12, 13, 14, 15);
}

// A fragment from LDS (16 rows valid, lda in elems, 16B-aligned rows)
static __device__ inline v16h load_a_lds(const _Float16* A, int lda, int k0) {
  int lane = threadIdx.x & 31;
  int m = lane & 15;
  int kh = (lane >> 4) & 1;
  const _Float16* p = A + m * lda + k0 + kh * 16;
  v8h lo = *(const v8h*)(p);
  v8h hi = *(const v8h*)(p + 8);
  return __builtin_shufflevector(lo, hi, 0, 1, 2, 3, 4, 5, 6, 7,
                                 8, 9, 10, 11, 12, 13, 14, 15);
}

// B fragment (32x16 f16 tile) from COLUMN-MAJOR weights Bt[N][K]:
// lane n=lane&15 holds column col0+n; elements = K k0+kg*16.. contiguous (32B).
static __device__ inline v16h load_bt_frag(const _Float16* Bt, int Kd,
                                           int k0, int col0) {
  int lane = threadIdx.x & 31;
  int n  = col0 + (lane & 15);
  int kg = (lane >> 4) & 1;
  return *(const v16h*)(Bt + (size_t)n * Kd + k0 + kg * 16);
}

// one wave computes a 32x32 output tile (2x2 WMMA tiles), K multiple of 32
static __device__ inline void wmma_tile_32x32(const _Float16* A, const _Float16* Bt,
                                              int M, int K, int row0, int col0,
                                              v8f& c00, v8f& c01, v8f& c10, v8f& c11) {
  c00 = v8f{}; c01 = v8f{}; c10 = v8f{}; c11 = v8f{};
  for (int k0 = 0; k0 < K; k0 += 32) {
    v16h a0 = load_a_frag(A, K, M, row0,      k0);
    v16h a1 = load_a_frag(A, K, M, row0 + 16, k0);
    v16h b0 = load_bt_frag(Bt, K, k0, col0);
    v16h b1 = load_bt_frag(Bt, K, k0, col0 + 16);
    c00 = WMMA_F16(a0, b0, c00);
    c01 = WMMA_F16(a0, b1, c01);
    c10 = WMMA_F16(a1, b0, c10);
    c11 = WMMA_F16(a1, b1, c11);
  }
}

// ---------------------------------------------------------------------------
// Problem constants
// ---------------------------------------------------------------------------
#define EMBED 1024
#define HEADS 16
#define HDIM  64
#define WIN   14
#define NTOK  196          // WIN*WIN
#define NPAD  224          // 14 tiles of 16
#define BW    50           // 2 images * 5*5 windows
#define MTOK  9800         // BW*NTOK
#define TOKS  8192         // 2*64*64 real tokens
#define MLPD  4096

// ---------------------------------------------------------------------------
// Prep: fold LoRA into QKV weight (W_eff = W + 2 * A@B), emit TRANSPOSED f16
// ---------------------------------------------------------------------------
__global__ void prep_wqkv_kernel(const float* __restrict__ qkv_w,
                                 const float* __restrict__ la,
                                 const float* __restrict__ lb,
                                 _Float16* __restrict__ wt) {   // [3072][1024]
  int i = blockIdx.x * 256 + threadIdx.x;
  if (i >= EMBED * 3 * EMBED) return;
  int k = i / (3 * EMBED), m = i % (3 * EMBED);
  float acc = qkv_w[i];
#pragma unroll
  for (int r = 0; r < 16; ++r) acc += 2.0f * la[k * 16 + r] * lb[r * 3 * EMBED + m];
  wt[(size_t)m * EMBED + k] = (_Float16)acc;
}

// transpose-convert f32 [K][N] row-major -> f16 [N][K]
__global__ void conv_t_kernel(const float* __restrict__ s, _Float16* __restrict__ d,
                              int K, int N) {
  int i = blockIdx.x * 256 + threadIdx.x;
  if (i >= K * N) return;
  int k = i / N, n = i % N;
  d[(size_t)n * K + k] = (_Float16)s[i];
}

__global__ void zero_u32_kernel(unsigned int* __restrict__ p, int n) {
  int i = blockIdx.x * 256 + threadIdx.x;
  if (i < n) p[i] = 0u;
}

// ---------------------------------------------------------------------------
// LayerNorm1 + window partition (zero-padded 14x14 windows) -> f16 rows
// ---------------------------------------------------------------------------
__global__ void ln1_win_kernel(const float* __restrict__ x,
                               const float* __restrict__ g,
                               const float* __restrict__ b,
                               _Float16* __restrict__ xw) {
  int t = blockIdx.x;                       // 0..MTOK-1
  int bwi = t / NTOK, pos = t % NTOK;
  int i = pos / WIN, j = pos % WIN;
  int img = bwi / 25, wr = (bwi % 25) / 5, wc = bwi % 5;
  int gi = wr * WIN + i, gj = wc * WIN + j;
  int tid = threadIdx.x;                    // 256 threads
  _Float16* dst = xw + (size_t)t * EMBED;
  if (gi >= 64 || gj >= 64) {
    for (int c = tid; c < EMBED; c += 256) dst[c] = (_Float16)0.f;
    return;
  }
  const float* src = x + (((size_t)(img * 64 + gi)) * 64 + gj) * EMBED;
  __shared__ float s1[256], s2[256];
  float a = 0.f, q = 0.f;
  for (int c = tid; c < EMBED; c += 256) { float v = src[c]; a += v; q += v * v; }
  s1[tid] = a; s2[tid] = q; __syncthreads();
  for (int off = 128; off > 0; off >>= 1) {
    if (tid < off) { s1[tid] += s1[tid + off]; s2[tid] += s2[tid + off]; }
    __syncthreads();
  }
  float mu = s1[0] * (1.f / EMBED);
  float var = s2[0] * (1.f / EMBED) - mu * mu;
  float rstd = rsqrtf(var + 1e-6f);
  for (int c = tid; c < EMBED; c += 256)
    dst[c] = (_Float16)((src[c] - mu) * rstd * g[c] + b[c]);
}

// LayerNorm2 (plain, on x1 rows) -> f16
__global__ void ln2_kernel(const float* __restrict__ x1,
                           const float* __restrict__ g,
                           const float* __restrict__ b,
                           _Float16* __restrict__ xn) {
  int t = blockIdx.x;                       // 0..TOKS-1
  int tid = threadIdx.x;
  const float* src = x1 + (size_t)t * EMBED;
  _Float16* dst = xn + (size_t)t * EMBED;
  __shared__ float s1[256], s2[256];
  float a = 0.f, q = 0.f;
  for (int c = tid; c < EMBED; c += 256) { float v = src[c]; a += v; q += v * v; }
  s1[tid] = a; s2[tid] = q; __syncthreads();
  for (int off = 128; off > 0; off >>= 1) {
    if (tid < off) { s1[tid] += s1[tid + off]; s2[tid] += s2[tid + off]; }
    __syncthreads();
  }
  float mu = s1[0] * (1.f / EMBED);
  float var = s2[0] * (1.f / EMBED) - mu * mu;
  float rstd = rsqrtf(var + 1e-6f);
  for (int c = tid; c < EMBED; c += 256)
    dst[c] = (_Float16)((src[c] - mu) * rstd * g[c] + b[c]);
}

// ---------------------------------------------------------------------------
// QKV GEMM: Xw(9800x1024) @ WeffT + bias -> q/k f16 (k pre-scaled), V transposed
// ---------------------------------------------------------------------------
__global__ void gemm_qkv_kernel(const _Float16* __restrict__ A,
                                const _Float16* __restrict__ Bt,
                                const float* __restrict__ bias,
                                _Float16* __restrict__ qb,
                                _Float16* __restrict__ kb,
                                _Float16* __restrict__ vt) {
  const int M = MTOK, K = EMBED;
  int tid = threadIdx.x, wave = tid >> 5, lane = tid & 31;
  int row0 = blockIdx.y * 64 + (wave >> 2) * 32;
  int col0 = blockIdx.x * 128 + (wave & 3) * 32;
  v8f c00, c01, c10, c11;
  wmma_tile_32x32(A, Bt, M, K, row0, col0, c00, c01, c10, c11);
  int n = lane & 15, g = (lane >> 4) & 1;
  auto emit = [&](v8f& c, int r0, int cc0) {
#pragma unroll
    for (int rr = 0; rr < 8; ++rr) {
      int row = r0 + g * 8 + rr;
      if (row >= M) continue;
      int col = cc0 + n;
      float val = c[rr] + bias[col];
      int which = col >> 10, rem = col & 1023;
      int head = rem >> 6, d = rem & 63;
      int bw = row / NTOK, nn = row % NTOK;
      if (which == 0) {
        qb[((size_t)(bw * HEADS + head) * NTOK + nn) * HDIM + d] = (_Float16)val;
      } else if (which == 1) {
        kb[((size_t)(bw * HEADS + head) * NTOK + nn) * HDIM + d] =
            (_Float16)(val * 0.125f);
      } else {                      // V stored transposed: [bw,h][d][key(pad 224)]
        vt[((size_t)(bw * HEADS + head) * HDIM + d) * NPAD + nn] = (_Float16)val;
      }
    }
  };
  emit(c00, row0, col0); emit(c01, row0, col0 + 16);
  emit(c10, row0 + 16, col0); emit(c11, row0 + 16, col0 + 16);
}

// ---------------------------------------------------------------------------
// Fused window attention per (qtile, head, window):
//   S = Q Kscaled^T (WMMA), decomposed rel-pos bias + mask, softmax, O = P V (WMMA)
// ---------------------------------------------------------------------------
__global__ void attn_kernel(const _Float16* __restrict__ qbuf,
                            const _Float16* __restrict__ kbuf,
                            const _Float16* __restrict__ vtbuf,
                            const float* __restrict__ relh,
                            const float* __restrict__ relw,
                            _Float16* __restrict__ aout) {
  int qt   = blockIdx.x;   // 0..12 query tile
  int head = blockIdx.y;   // 0..15
  int bw   = blockIdx.z;   // 0..49
  const _Float16* Q  = qbuf  + ((size_t)(bw * HEADS + head)) * NTOK * HDIM;
  const _Float16* Kk = kbuf  + ((size_t)(bw * HEADS + head)) * NTOK * HDIM;
  const _Float16* Vt = vtbuf + ((size_t)(bw * HEADS + head)) * HDIM * NPAD;

  __shared__ _Float16 Qs[16 * HDIM];        //  2 KB
  __shared__ float    S[16 * NPAD];         // 14 KB
  __shared__ _Float16 P[16 * NPAD];         //  7 KB
  __shared__ float    RelH[16][WIN];        // 896 B
  __shared__ float    RelW[16][WIN];        // 896 B

  int tid = threadIdx.x;                    // 128 threads = 4 waves
  int lane = tid & 31, wave = tid >> 5;
  int q0 = qt * 16;

  // stage Q tile to LDS (clamped rows; clamped rows never stored)
  for (int idx = tid; idx < 16 * HDIM; idx += 128) {
    int r = idx >> 6, d = idx & 63;
    int qg = q0 + r; if (qg > NTOK - 1) qg = NTOK - 1;
    Qs[idx] = Q[qg * HDIM + d];
  }
  __syncthreads();

  // S = Q @ K^T  (14 key tiles over 4 waves, K-dim = 64)
  for (int j = wave; j < NPAD / 16; j += 4) {
    v8f c = v8f{};
#pragma unroll
    for (int kh = 0; kh < 2; ++kh) {
      v16h a = load_a_lds(Qs, HDIM, kh * 32);
      int key = j * 16 + (lane & 15); if (key > NTOK - 1) key = NTOK - 1;
      int kg = (lane >> 4) & 1;
      v16h b = *(const v16h*)(Kk + (size_t)key * HDIM + kh * 32 + kg * 16);
      c = WMMA_F16(a, b, c);
    }
    int n = lane & 15, g = (lane >> 4) & 1;
#pragma unroll
    for (int rr = 0; rr < 8; ++rr) S[(g * 8 + rr) * NPAD + j * 16 + n] = c[rr];
  }

  // decomposed rel-pos projections: rel_h[r][kx], rel_w[r][ky]
  for (int idx = tid; idx < 2 * 16 * WIN; idx += 128) {
    int sel = idx / (16 * WIN);             // 0 -> H, 1 -> W
    int rem = idx % (16 * WIN);
    int r = rem / WIN, kk = rem % WIN;
    int qg = q0 + r; if (qg > NTOK - 1) qg = NTOK - 1;
    int qc = sel ? (qg % WIN) : (qg / WIN);
    const float* tab = (sel ? relw : relh) + (qc - kk + WIN - 1) * HDIM;
    float acc = 0.f;
#pragma unroll 8
    for (int d = 0; d < HDIM; ++d) acc += (float)Qs[r * HDIM + d] * tab[d];
    if (sel) RelW[r][kk] = acc; else RelH[r][kk] = acc;
  }
  __syncthreads();

  // add bias, mask padded keys / padded queries
  for (int idx = tid; idx < 16 * NPAD; idx += 128) {
    int r = idx / NPAD, cc = idx % NPAD;
    int qg = q0 + r;
    if (cc >= NTOK || qg >= NTOK) S[idx] = -1e30f;
    else S[idx] += RelH[r][cc / WIN] + RelW[r][cc % WIN];
  }
  __syncthreads();

  // softmax over keys (one thread per row)
  if (tid < 16) {
    float mx = -1e30f;
    for (int c2 = 0; c2 < NPAD; ++c2) mx = fmaxf(mx, S[tid * NPAD + c2]);
    float sum = 0.f;
    for (int c2 = 0; c2 < NPAD; ++c2) {
      float e = __expf(S[tid * NPAD + c2] - mx);
      S[tid * NPAD + c2] = e; sum += e;
    }
    float inv = 1.f / sum;
    for (int c2 = 0; c2 < NPAD; ++c2)
      P[tid * NPAD + c2] = (_Float16)(S[tid * NPAD + c2] * inv);
  }
  __syncthreads();

  // O = P(16x224) @ V(224x64): one 16-wide d-tile per wave, 7 K steps
  {
    int nt = wave; // 0..3
    v8f c = v8f{};
    for (int k0 = 0; k0 < NPAD; k0 += 32) {
      v16h a = load_a_lds(P, NPAD, k0);
      // Vt is [d][key], key padded to 224 with zeros -> contiguous 32B per lane
      int n = lane & 15, kg = (lane >> 4) & 1;
      v16h b = *(const v16h*)(Vt + (size_t)(nt * 16 + n) * NPAD + k0 + kg * 16);
      c = WMMA_F16(a, b, c);
    }
    int n = lane & 15, g = (lane >> 4) & 1;
#pragma unroll
    for (int rr = 0; rr < 8; ++rr) {
      int qg = q0 + g * 8 + rr;
      if (qg < NTOK) {
        size_t token = (size_t)bw * NTOK + qg;
        aout[token * EMBED + head * HDIM + nt * 16 + n] = (_Float16)c[rr];
      }
    }
  }
}

// ---------------------------------------------------------------------------
// Proj GEMM + window unpartition + residual: x1 = x + unpart(AO @ Wp + bp)
// ---------------------------------------------------------------------------
__global__ void gemm_proj_kernel(const _Float16* __restrict__ A,
                                 const _Float16* __restrict__ Bt,
                                 const float* __restrict__ bias,
                                 const float* __restrict__ xin,
                                 float* __restrict__ x1) {
  const int M = MTOK, K = EMBED;
  int tid = threadIdx.x, wave = tid >> 5, lane = tid & 31;
  int row0 = blockIdx.y * 64 + (wave >> 2) * 32;
  int col0 = blockIdx.x * 128 + (wave & 3) * 32;
  v8f c00, c01, c10, c11;
  wmma_tile_32x32(A, Bt, M, K, row0, col0, c00, c01, c10, c11);
  int n = lane & 15, g = (lane >> 4) & 1;
  auto emit = [&](v8f& c, int r0, int cc0) {
#pragma unroll
    for (int rr = 0; rr < 8; ++rr) {
      int row = r0 + g * 8 + rr;
      if (row >= M) continue;
      int col = cc0 + n;
      int bw = row / NTOK, pos = row % NTOK;
      int i = pos / WIN, j = pos % WIN;
      int img = bw / 25, wr = (bw % 25) / 5, wc = bw % 5;
      int gi = wr * WIN + i, gj = wc * WIN + j;
      if (gi < 64 && gj < 64) {
        size_t idx = (((size_t)(img * 64 + gi)) * 64 + gj) * EMBED + col;
        x1[idx] = c[rr] + bias[col] + xin[idx];
      }
    }
  };
  emit(c00, row0, col0); emit(c01, row0, col0 + 16);
  emit(c10, row0 + 16, col0); emit(c11, row0 + 16, col0 + 16);
}

// ---------------------------------------------------------------------------
// MLP GEMM 1: xn2 @ W1t + b1 -> exact GELU -> hmid (f16)
// ---------------------------------------------------------------------------
__global__ void gemm_mlp1_kernel(const _Float16* __restrict__ A,
                                 const _Float16* __restrict__ Bt,
                                 const float* __restrict__ bias,
                                 _Float16* __restrict__ hmid) {
  const int M = TOKS, K = EMBED;
  int tid = threadIdx.x, wave = tid >> 5, lane = tid & 31;
  int row0 = blockIdx.y * 64 + (wave >> 2) * 32;
  int col0 = blockIdx.x * 128 + (wave & 3) * 32;
  v8f c00, c01, c10, c11;
  wmma_tile_32x32(A, Bt, M, K, row0, col0, c00, c01, c10, c11);
  int n = lane & 15, g = (lane >> 4) & 1;
  auto emit = [&](v8f& c, int r0, int cc0) {
#pragma unroll
    for (int rr = 0; rr < 8; ++rr) {
      int row = r0 + g * 8 + rr;
      if (row >= M) continue;
      int col = cc0 + n;
      float t = c[rr] + bias[col];
      float ge = 0.5f * t * (1.0f + erff(t * 0.70710678118654752f)); // exact GELU
      hmid[(size_t)row * MLPD + col] = (_Float16)ge;
    }
  };
  emit(c00, row0, col0); emit(c01, row0, col0 + 16);
  emit(c10, row0 + 16, col0); emit(c11, row0 + 16, col0 + 16);
}

// ---------------------------------------------------------------------------
// MLP GEMM 2: hmid @ W2t + b2 + x1 -> d_out (f32)
// ---------------------------------------------------------------------------
__global__ void gemm_mlp2_kernel(const _Float16* __restrict__ A,
                                 const _Float16* __restrict__ Bt,
                                 const float* __restrict__ bias,
                                 const float* __restrict__ x1,
                                 float* __restrict__ out) {
  const int M = TOKS, K = MLPD;
  int tid = threadIdx.x, wave = tid >> 5, lane = tid & 31;
  int row0 = blockIdx.y * 64 + (wave >> 2) * 32;
  int col0 = blockIdx.x * 128 + (wave & 3) * 32;
  v8f c00, c01, c10, c11;
  wmma_tile_32x32(A, Bt, M, K, row0, col0, c00, c01, c10, c11);
  int n = lane & 15, g = (lane >> 4) & 1;
  auto emit = [&](v8f& c, int r0, int cc0) {
#pragma unroll
    for (int rr = 0; rr < 8; ++rr) {
      int row = r0 + g * 8 + rr;
      if (row >= M) continue;
      int col = cc0 + n;
      size_t idx = (size_t)row * EMBED + col;
      out[idx] = c[rr] + bias[col] + x1[idx];
    }
  };
  emit(c00, row0, col0); emit(c01, row0, col0 + 16);
  emit(c10, row0 + 16, col0); emit(c11, row0 + 16, col0 + 16);
}

// ---------------------------------------------------------------------------
// Host launch
// ---------------------------------------------------------------------------
extern "C" void kernel_launch(void* const* d_in, const int* in_sizes, int n_in,
                              void* d_out, int out_size, void* d_ws, size_t ws_size,
                              hipStream_t stream) {
  (void)in_sizes; (void)n_in; (void)out_size; (void)ws_size;
  const float* x      = (const float*)d_in[0];
  const float* n1g    = (const float*)d_in[1];
  const float* n1b    = (const float*)d_in[2];
  const float* qkv_w  = (const float*)d_in[3];
  const float* qkv_b  = (const float*)d_in[4];
  const float* lora_a = (const float*)d_in[5];
  const float* lora_b = (const float*)d_in[6];
  const float* relh   = (const float*)d_in[7];
  const float* relw   = (const float*)d_in[8];
  const float* proj_w = (const float*)d_in[9];
  const float* proj_b = (const float*)d_in[10];
  const float* n2g    = (const float*)d_in[11];
  const float* n2b    = (const float*)d_in[12];
  const float* w1     = (const float*)d_in[13];
  const float* b1     = (const float*)d_in[14];
  const float* w2     = (const float*)d_in[15];
  const float* b2     = (const float*)d_in[16];
  float* out = (float*)d_out;

  // workspace layout (bytes); xn2 aliases xw, hmid aliases q/k region
  char* ws = (char*)d_ws;
  _Float16* wqkv_t  = (_Float16*)(ws + 0);          // 3072*1024*2 = 6,291,456
  _Float16* projw_t = (_Float16*)(ws + 6291456);    // 1024*1024*2 = 2,097,152
  _Float16* w1_t    = (_Float16*)(ws + 8388608);    // 4096*1024*2 = 8,388,608
  _Float16* w2_t    = (_Float16*)(ws + 16777216);   // 1024*4096*2 = 8,388,608
  float*    x1      = (float*)   (ws + 25165824);   // 8192*1024*4 = 33,554,432
  _Float16* xw_h    = (_Float16*)(ws + 58720256);   // 9800*1024*2 = 20,070,400
  _Float16* qbuf    = (_Float16*)(ws + 78790656);   // 20,070,400
  _Float16* kbuf    = (_Float16*)(ws + 98861056);   // 20,070,400
  _Float16* vtbuf   = (_Float16*)(ws + 118931456);  // 50*16*64*224*2 = 22,937,600
  _Float16* aobuf   = (_Float16*)(ws + 141869056);  // 20,070,400 (end 161,939,456)
  _Float16* xn2_h   = xw_h;                         // alias (LN2 after attn done)
  _Float16* hmid_h  = qbuf;                         // alias (q/k/vt/ao dead by MLP)

  // 1. fold LoRA into QKV weight; transpose-convert all weights to f16 [N][K]
  prep_wqkv_kernel<<<(EMBED * 3 * EMBED + 255) / 256, 256, 0, stream>>>(
      qkv_w, lora_a, lora_b, wqkv_t);
  conv_t_kernel<<<(EMBED * EMBED + 255) / 256, 256, 0, stream>>>(
      proj_w, projw_t, EMBED, EMBED);
  conv_t_kernel<<<(EMBED * MLPD + 255) / 256, 256, 0, stream>>>(
      w1, w1_t, EMBED, MLPD);
  conv_t_kernel<<<(MLPD * EMBED + 255) / 256, 256, 0, stream>>>(
      w2, w2_t, MLPD, EMBED);

  // 2. LN1 + window partition; zero V-transposed buffer (key padding)
  ln1_win_kernel<<<MTOK, 256, 0, stream>>>(x, n1g, n1b, xw_h);
  zero_u32_kernel<<<(int)((22937600 / 4 + 255) / 256), 256, 0, stream>>>(
      (unsigned int*)vtbuf, 22937600 / 4);

  // 3. QKV GEMM (WMMA) -> q/k (k pre-scaled), V transposed+padded
  gemm_qkv_kernel<<<dim3(3 * EMBED / 128, (MTOK + 63) / 64), 256, 0, stream>>>(
      xw_h, wqkv_t, qkv_b, qbuf, kbuf, vtbuf);

  // 4. fused window attention (WMMA for QK^T and PV)
  attn_kernel<<<dim3(NPAD / 16 - 1, HEADS, BW), 128, 0, stream>>>(
      qbuf, kbuf, vtbuf, relh, relw, aobuf);

  // 5. proj GEMM + unpartition + residual
  gemm_proj_kernel<<<dim3(EMBED / 128, (MTOK + 63) / 64), 256, 0, stream>>>(
      aobuf, projw_t, proj_b, x, x1);

  // 6. LN2
  ln2_kernel<<<TOKS, 256, 0, stream>>>(x1, n2g, n2b, xn2_h);

  // 7. MLP1 GEMM + exact GELU
  gemm_mlp1_kernel<<<dim3(MLPD / 128, TOKS / 64), 256, 0, stream>>>(
      xn2_h, w1_t, b1, hmid_h);

  // 8. MLP2 GEMM + residual -> out
  gemm_mlp2_kernel<<<dim3(EMBED / 128, TOKS / 64), 256, 0, stream>>>(
      hmid_h, w2_t, b2, x1, out);
}
